// ReCDR_91216515432632
// MI455X (gfx1250) — matrix operand order repaired and checked by reading, not compiled, not run on hardware
//
#include <hip/hip_runtime.h>
#include <math.h>

typedef __attribute__((ext_vector_type(16))) _Float16 v16h_t;
typedef __attribute__((ext_vector_type(8)))  float    v8f_t;

// ---------------------------------------------------------------------------
// fill
// ---------------------------------------------------------------------------
__global__ void k_fill(float* __restrict__ p, int n, float v) {
  int i = blockIdx.x * blockDim.x + threadIdx.x;
  if (i < n) p[i] = v;
}

// ---------------------------------------------------------------------------
// GEMM: out = X[V,64] @ W[64,64]  via v_wmma_f32_16x16x32_f16
// MODE 0: out[v*64+c]        = tanh(acc)
// MODE 1: out[v*128+64+c]    = Ws[v*64+c]*acc + out[v*128+64+c]   (S = Ws*Fg + Np)
// One wave computes a 16x64 tile: 2 k-steps x 4 col-tiles = 8 WMMAs.
// ---------------------------------------------------------------------------
template<int MODE>
__global__ void k_gemm64(const float* __restrict__ X, const float* __restrict__ W,
                         float* __restrict__ out, const float* __restrict__ Ws, int V)
{
  int wid  = (blockIdx.x * blockDim.x + threadIdx.x) >> 5;
  int lane = threadIdx.x & 31;
  int row0 = wid * 16;
  if (row0 >= V) return;                  // uniform per wave; EXEC stays all-1s
  const int half = lane >> 4;             // lane group 0..15 / 16..31
  const int l15  = lane & 15;

  // ---- A fragments (16x32 f16, MxK): lane holds row M=l15;
  //      vgpr j -> K = half*16 + 2j (+1)
  const float* xrow = X + (long)(row0 + l15) * 64;
  v16h_t a0, a1;
#pragma unroll
  for (int j = 0; j < 8; ++j) {
    int k0 = half * 16 + 2 * j;
    a0[2 * j]     = (_Float16)xrow[k0];
    a0[2 * j + 1] = (_Float16)xrow[k0 + 1];
    a1[2 * j]     = (_Float16)xrow[32 + k0];
    a1[2 * j + 1] = (_Float16)xrow[32 + k0 + 1];
  }

  v8f_t acc[4] = {};
#pragma unroll
  for (int s = 0; s < 2; ++s) {
    v16h_t a = s ? a1 : a0;
#pragma unroll
    for (int t = 0; t < 4; ++t) {
      // ---- B fragment (32x16 f16, KxN): lane holds col N=l15;
      //      lanes 0-15 hold K=0..15, lanes 16-31 hold K=16..31 (2 per vgpr)
      v16h_t b;
      int n = t * 16 + l15;
#pragma unroll
      for (int j = 0; j < 8; ++j) {
        int k = s * 32 + half * 16 + 2 * j;
        b[2 * j]     = (_Float16)W[k * 64 + n];
        b[2 * j + 1] = (_Float16)W[(k + 1) * 64 + n];
      }
      acc[t] = __builtin_amdgcn_wmma_f32_16x16x32_f16(
          /*neg_a=*/false, a, /*neg_b=*/false, b,
          /*c_mod=*/(short)0, acc[t], /*reuse_a=*/false, /*reuse_b=*/false);
    }
  }

  // ---- epilogue: C layout: vgpr r -> row M = r + half*8, col N = l15 (per tile)
#pragma unroll
  for (int t = 0; t < 4; ++t) {
#pragma unroll
    for (int r = 0; r < 8; ++r) {
      long rr = row0 + r + half * 8;
      int  cc = t * 16 + l15;
      float v = acc[t][r];
      if (MODE == 0) {
        out[rr * 64 + cc] = tanhf(v);
      } else {
        float np = out[rr * 128 + 64 + cc];        // Np staged here earlier
        out[rr * 128 + 64 + cc] = Ws[rr * 64 + cc] * v + np;
      }
    }
  }
}

// ---------------------------------------------------------------------------
// e[v] = dot(T[v,:], w[0:64])  -- one wave per row, shuffle reduction
// ---------------------------------------------------------------------------
__global__ void k_rowdot(const float* __restrict__ T, const float* __restrict__ w,
                         float* __restrict__ e, int V)
{
  int wid  = (blockIdx.x * blockDim.x + threadIdx.x) >> 5;
  int lane = threadIdx.x & 31;
  if (wid >= V) return;
  const float* row = T + (long)wid * 64;
  float s = row[lane] * w[lane] + row[lane + 32] * w[lane + 32];
#pragma unroll
  for (int off = 16; off > 0; off >>= 1) s += __shfl_xor(s, off, 32);
  if (lane == 0) e[wid] = s;
}

// ---------------------------------------------------------------------------
// segment max:  m[dst] = max(m[dst], e[src])   (m pre-filled with -inf)
// ---------------------------------------------------------------------------
__global__ void k_edge_max(const int* __restrict__ src, const int* __restrict__ dst,
                           int ne, const float* __restrict__ e, float* __restrict__ m)
{
  int i = blockIdx.x * blockDim.x + threadIdx.x;
  if (i < ne) atomicMax(&m[dst[i]], e[src[i]]);
}

// ---------------------------------------------------------------------------
// softmax scatter: w = exp(e[src]-m[dst]); denom[dst]+=w; agg[dst,:] += w*N[src,:]
// one wave per edge; lane covers cols {lane, lane+32}
// ---------------------------------------------------------------------------
__global__ void k_softmax_scatter(const int* __restrict__ src, const int* __restrict__ dst,
                                  int ne, const float* __restrict__ e,
                                  const float* __restrict__ m, const float* __restrict__ N,
                                  float* __restrict__ agg, float* __restrict__ denom)
{
  int wid  = (blockIdx.x * blockDim.x + threadIdx.x) >> 5;
  int lane = threadIdx.x & 31;
  if (wid >= ne) return;
  int s = src[wid], d = dst[wid];
  float w = __expf(e[s] - m[d]);
  if (lane == 0) atomicAdd(&denom[d], w);
  const float* nr = N + (long)s * 64;
  float* ar = agg + (long)d * 64;
  atomicAdd(&ar[lane],      w * nr[lane]);
  atomicAdd(&ar[lane + 32], w * nr[lane + 32]);
}

// ---------------------------------------------------------------------------
// mean scatter: cnt[dst] += 1; Y[dst,:] += E[edge,:]
// ---------------------------------------------------------------------------
__global__ void k_mean_scatter(const int* __restrict__ dst, int ne,
                               const float* __restrict__ E,
                               float* __restrict__ Y, float* __restrict__ cnt)
{
  int wid  = (blockIdx.x * blockDim.x + threadIdx.x) >> 5;
  int lane = threadIdx.x & 31;
  if (wid >= ne) return;
  int d = dst[wid];
  if (lane == 0) atomicAdd(&cnt[d], 1.0f);
  const float* er = E + (long)wid * 64;
  float* yr = Y + (long)d * 64;
  atomicAdd(&yr[lane],      er[lane]);
  atomicAdd(&yr[lane + 32], er[lane + 32]);
}

// ---------------------------------------------------------------------------
// out[v*rowStride + colOff + c] = acc[v*64+c] / (den[v] > 0 ? den[v] : 1)
// ---------------------------------------------------------------------------
__global__ void k_scale_div(const float* __restrict__ acc, const float* __restrict__ den,
                            float* __restrict__ out, int rowStride, int colOff, int V)
{
  int i = blockIdx.x * blockDim.x + threadIdx.x;
  if (i >= V * 64) return;
  int v = i >> 6, c = i & 63;
  float d = den[v];
  out[(long)v * rowStride + colOff + c] = acc[i] / (d > 0.f ? d : 1.f);
}

// ---------------------------------------------------------------------------
// 2-way attention combine: YI := a0*YI + a1*YS, a = softmax([eI, eS])
// ---------------------------------------------------------------------------
__global__ void k_combine(float* __restrict__ YI, const float* __restrict__ YS,
                          const float* __restrict__ eI, const float* __restrict__ eS, int V)
{
  int i = blockIdx.x * blockDim.x + threadIdx.x;
  if (i >= V * 64) return;
  int v = i >> 6;
  float a = eI[v], b = eS[v];
  float mx = fmaxf(a, b);
  float wa = __expf(a - mx), wb = __expf(b - mx);
  float inv = 1.0f / (wa + wb);
  YI[i] = (wa * YI[i] + wb * YS[i]) * inv;
}

// ===========================================================================
extern "C" void kernel_launch(void* const* d_in, const int* in_sizes, int n_in,
                              void* d_out, int out_size, void* d_ws, size_t ws_size,
                              hipStream_t stream)
{
  constexpr int V       = 100000;
  constexpr int E_NG    = 2000000;
  constexpr int E_LOCAL = 1000000;
  constexpr int E_INT   = 1200000;
  constexpr int E_SIM   = 800000;

  const float* local_N  = (const float*)d_in[0];
  const float* global_N = (const float*)d_in[1];
  const float* edge_E   = (const float*)d_in[2];
  const float* Ws       = (const float*)d_in[3];
  const float* lW1      = (const float*)d_in[4];
  const float* lw2      = (const float*)d_in[5];
  const float* gW1      = (const float*)d_in[6];
  const float* gw2      = (const float*)d_in[7];
  const float* eW1      = (const float*)d_in[8];
  const float* ew2      = (const float*)d_in[9];
  const float* eW3      = (const float*)d_in[10];
  const int* ng_src     = (const int*)d_in[11];
  const int* ng_dst     = (const int*)d_in[12];
  const int* local_src  = (const int*)d_in[13];
  const int* local_dst  = (const int*)d_in[14];
  const int* int_dst    = (const int*)d_in[16];
  const int* sim_dst    = (const int*)d_in[18];

  float* out = (float*)d_out;             // H [V,128] flat (= users then items)

  // workspace layout (floats): 197*V total ~ 79 MB
  float* ws   = (float*)d_ws;
  float* buf0 = ws;                       // [V,64] : tanh-GEMM output, then agg accum
  float* YI   = ws + (long)64 * V;        // [V,64]
  float* YS   = ws + (long)128 * V;       // [V,64]
  float* evec = ws + (long)192 * V;       // [V] attention scores (reused per graph)
  float* mbuf = ws + (long)193 * V;       // [V] segment max
  float* den  = ws + (long)194 * V;       // [V] denom / counts (reused)
  float* eI   = ws + (long)195 * V;       // [V]
  float* eS   = ws + (long)196 * V;       // [V]

  const int TPB = 256;
  auto blocks = [](long threads) { return (int)((threads + 255) / 256); };
  const int gemmGrid = blocks((long)(V / 16) * 32);  // wave per 16-row tile
  const int rowGrid  = blocks((long)V * 32);         // wave per row
  const int elemGrid = blocks((long)V * 64);
  const int vGrid    = blocks(V);

  // ===================== local GAT =====================
  k_gemm64<0><<<gemmGrid, TPB, 0, stream>>>(local_N, lW1, buf0, nullptr, V);
  k_rowdot<<<rowGrid, TPB, 0, stream>>>(buf0, lw2, evec, V);
  k_fill<<<vGrid, TPB, 0, stream>>>(mbuf, V, -INFINITY);
  k_fill<<<vGrid, TPB, 0, stream>>>(den, V, 0.f);
  k_fill<<<elemGrid, TPB, 0, stream>>>(buf0, V * 64, 0.f);
  k_edge_max<<<blocks(E_LOCAL), TPB, 0, stream>>>(local_src, local_dst, E_LOCAL, evec, mbuf);
  k_softmax_scatter<<<blocks((long)E_LOCAL * 32), TPB, 0, stream>>>(
      local_src, local_dst, E_LOCAL, evec, mbuf, local_N, buf0, den);
  // local_Np -> H[:, 0:64]
  k_scale_div<<<elemGrid, TPB, 0, stream>>>(buf0, den, out, 128, 0, V);

  // ===================== global GAT =====================
  k_gemm64<0><<<gemmGrid, TPB, 0, stream>>>(global_N, gW1, buf0, nullptr, V);
  k_rowdot<<<rowGrid, TPB, 0, stream>>>(buf0, gw2, evec, V);
  k_fill<<<vGrid, TPB, 0, stream>>>(mbuf, V, -INFINITY);
  k_fill<<<vGrid, TPB, 0, stream>>>(den, V, 0.f);
  k_fill<<<elemGrid, TPB, 0, stream>>>(buf0, V * 64, 0.f);
  k_edge_max<<<blocks(E_NG), TPB, 0, stream>>>(ng_src, ng_dst, E_NG, evec, mbuf);
  k_softmax_scatter<<<blocks((long)E_NG * 32), TPB, 0, stream>>>(
      ng_src, ng_dst, E_NG, evec, mbuf, global_N, buf0, den);
  // Np staged in H[:, 64:128]; fused final GEMM reads it back
  k_scale_div<<<elemGrid, TPB, 0, stream>>>(buf0, den, out, 128, 64, V);

  // ===================== mean aggregations =====================
  k_fill<<<elemGrid, TPB, 0, stream>>>(YI, V * 64, 0.f);
  k_fill<<<vGrid, TPB, 0, stream>>>(den, V, 0.f);
  k_mean_scatter<<<blocks((long)E_INT * 32), TPB, 0, stream>>>(int_dst, E_INT, edge_E, YI, den);
  k_scale_div<<<elemGrid, TPB, 0, stream>>>(YI, den, YI, 64, 0, V);

  k_fill<<<elemGrid, TPB, 0, stream>>>(YS, V * 64, 0.f);
  k_fill<<<vGrid, TPB, 0, stream>>>(den, V, 0.f);
  k_mean_scatter<<<blocks((long)E_SIM * 32), TPB, 0, stream>>>(
      sim_dst, E_SIM, edge_E + (long)E_INT * 64, YS, den);
  k_scale_div<<<elemGrid, TPB, 0, stream>>>(YS, den, YS, 64, 0, V);

  // ===================== edge attention =====================
  k_gemm64<0><<<gemmGrid, TPB, 0, stream>>>(YI, eW1, buf0, nullptr, V);
  k_rowdot<<<rowGrid, TPB, 0, stream>>>(buf0, ew2, eI, V);
  k_gemm64<0><<<gemmGrid, TPB, 0, stream>>>(YS, eW1, buf0, nullptr, V);
  k_rowdot<<<rowGrid, TPB, 0, stream>>>(buf0, ew2, eS, V);
  k_combine<<<elemGrid, TPB, 0, stream>>>(YI, YS, eI, eS, V);   // YI := OY

  // ===================== final: H[:,64:128] = Ws*(OY@eW3) + Np =====================
  k_gemm64<1><<<gemmGrid, TPB, 0, stream>>>(YI, eW3, out, Ws, V);
}
